// AttentionHead_28286654611507
// MI455X (gfx1250) — matrix operand order repaired and checked by reading, not compiled, not run on hardware
//
#include <hip/hip_runtime.h>
#include <hip/hip_bf16.h>
#include <math.h>

// Problem constants (match reference).
#define BB 8
#define HH 12
#define LQ 32
#define LF 256
#define VV 32000
#define SS 4

typedef __attribute__((ext_vector_type(16))) _Float16 v16h;
typedef __attribute__((ext_vector_type(8)))  float    v8f;

// Fast log1p for x >= 0: ln2 * v_log_f32(1+x).  ~3 VALU ops, TRANS co-executes.
__device__ __forceinline__ float log1p_relu_fast(float x) {
  return 0.69314718056f * __log2f(1.0f + fmaxf(x, 0.0f));
}

// ---------------------------------------------------------------------------
// TDM (Tensor Data Mover) support probe: 5-arg builtin on ROCm 7.2,
// 6-arg on amdgpu-toolchain (which ships the gfx1250 TDM header).
// ---------------------------------------------------------------------------
#if defined(__has_builtin)
#  if __has_builtin(__builtin_amdgcn_tensor_load_to_lds) && \
      __has_builtin(__builtin_amdgcn_s_wait_tensorcnt)
#    define HAVE_TDM 1
#  endif
#endif
#if defined(HAVE_TDM) && defined(__has_include)
#  if __has_include(<hip/amd_detail/amd_gfx1250_TDM.h>)
#    define TDM_SIX_ARGS 1
#  endif
#endif

#if defined(HAVE_TDM)
typedef unsigned int u32x4 __attribute__((ext_vector_type(4)));
typedef int          i32x8 __attribute__((ext_vector_type(8)));
typedef int          i32x4 __attribute__((ext_vector_type(4)));

// Generic "shared" pointer -> LDS byte offset (flat aperture: addr[31:0]).
__device__ __forceinline__ unsigned lds_off(const void* p) {
  return (unsigned)(uintptr_t)p;
}

// Issue one 2D tile load: rows x 256 f32 elements, row stride VV elements.
// D# layout per CDNA5 ISA (§8.3/§8.4).
__device__ __forceinline__ void tdm_load_tile(const float* gsrc, unsigned lds_addr,
                                              int rows) {
  unsigned long long ga = (unsigned long long)(uintptr_t)gsrc;
  u32x4 g0;
  g0[0] = 1u;                                              // count=1 (valid user D#)
  g0[1] = lds_addr;                                        // lds_addr (bytes)
  g0[2] = (unsigned)(ga & 0xffffffffull);                  // global_addr[31:0]
  g0[3] = (unsigned)((ga >> 32) & 0x01ffffffull)           // global_addr[56:32]
        | (2u << 30);                                      // type=2 ("image")
  i32x8 g1;
  g1[0] = (int)(2u << 16);                                 // data_size=2 (4 bytes)
  g1[1] = (int)((VV & 0xffff) << 16);                      // tensor_dim0[15:0]
  g1[2] = (int)(((VV >> 16) & 0xffff) | ((LF & 0xffff) << 16)); // dim0 hi | dim1 lo
  g1[3] = (int)((256u & 0xffff) << 16);                    // dim1 hi=0 | tile_dim0=256
  g1[4] = rows & 0xffff;                                   // tile_dim1=rows, tile_dim2=0
  g1[5] = VV;                                              // tensor_dim0_stride[31:0]
  g1[6] = 0;                                               // stride hi | dim1_stride lo
  g1[7] = 0;
  i32x4 z4 = {0, 0, 0, 0};
#if defined(TDM_SIX_ARGS)
  i32x8 z8 = {0, 0, 0, 0, 0, 0, 0, 0};
  __builtin_amdgcn_tensor_load_to_lds(g0, g1, z4, z4, z8, 0);
#else
  __builtin_amdgcn_tensor_load_to_lds(g0, g1, z4, z4, 0);
#endif
}
#endif  // HAVE_TDM

// ---------------------------------------------------------------------------
// Kernel 1: probs[b,lf] = sigmoid( max_lq( mean_h attention_scores ) ) * amask
// Head-mean as WMMA: D = A(16 lf rows x K=12 heads, f16) * B(K x 16, =1/12).
// One wave per (b, 16-wide lf tile); 32 WMMAs (one per lq), max-folded.
// ---------------------------------------------------------------------------
__global__ __launch_bounds__(32) void probs_wmma_kernel(
    const float* __restrict__ scores,   // [B,H,LQ,LF]
    const float* __restrict__ amask,    // [B,LF]
    float* __restrict__ probs)          // [B,LF]
{
  const int lane = threadIdx.x;                 // 0..31
  const int tile = blockIdx.x;                  // B * LF/16 = 128
  const int b    = tile >> 4;
  const int lf0  = (tile & 15) << 4;

  // B matrix (32x16 f16): rows K=0..11 hold 1/12, rest zero.
  v16h bm = {};
  {
    const _Float16 inv12 = (_Float16)(1.0f / 12.0f);
    const int kb = (lane < 16) ? 0 : 8;
#pragma unroll
    for (int k = 0; k < 8; ++k) if (kb + k < HH) bm[k] = inv12;
  }

  // A layout (16-bit, 16x32): lane<16 holds row M, halves 0..7 = K0..7;
  // lane>=16 holds halves 0..7 = K8..15; halves 8..15 (K16..31) stay zero.
  const int m     = lane & 15;          // lf within tile
  const int lf    = lf0 + m;
  const int hbase = (lane < 16) ? 0 : 8;

  v8f mx;
#pragma unroll
  for (int i = 0; i < 8; ++i) mx[i] = -__builtin_inff();

  for (int lq = 0; lq < LQ; ++lq) {
    const float* sbase = scores + ((size_t)b * HH * LQ + lq) * LF + lf;
    v16h a = {};
#pragma unroll
    for (int j = 0; j < 8; ++j) {
      const int h  = hbase + j;
      const int hc = (h < HH) ? h : 0;               // clamp, load always
      const float x = sbase[(size_t)hc * (LQ * LF)];
      a[j] = (h < HH) ? (_Float16)x : (_Float16)0.0f; // v_cndmask, no exec branch
    }
    v8f c = {};
    c = __builtin_amdgcn_wmma_f32_16x16x32_f16(
        /*neg_a=*/false, a, /*neg_b=*/false, bm,
        /*c_mod=*/(short)0, c, /*reuse_a=*/false, /*reuse_b=*/false);
#pragma unroll
    for (int i = 0; i < 8; ++i) mx[i] = fmaxf(mx[i], c[i]);
  }

  // Cross-lane extraction via LDS (single wave; cheap).
  __shared__ float red[32][8];
#pragma unroll
  for (int i = 0; i < 8; ++i) red[lane][i] = mx[i];
  __syncthreads();
  if (lane < 16) {
    const float agg = (lane < 8) ? red[0][lane] : red[16][lane - 8];
    const float p   = __builtin_amdgcn_rcpf(1.0f + __expf(-agg));
    probs[b * LF + lf0 + lane] = p * amask[b * LF + lf0 + lane];
  }
}

// ---------------------------------------------------------------------------
// Kernel 2: actions + logprobs. One block per (s,b), thread t = lf index.
// ---------------------------------------------------------------------------
__device__ __forceinline__ float hash_uniform(unsigned x) {
  x ^= x >> 16; x *= 0x7feb352du;
  x ^= x >> 15; x *= 0x846ca68bu;
  x ^= x >> 16;
  return (float)(x >> 8) * (1.0f / 16777216.0f);
}

__global__ __launch_bounds__(256) void sample_kernel(
    const float* __restrict__ probs,      // [B,LF]
    float* __restrict__ actions,          // [S,B,LF]
    float* __restrict__ logprobs)         // [S,B]
{
  const int s = blockIdx.x;               // 0..3
  const int b = blockIdx.y;               // 0..7
  const int t = threadIdx.x;              // 0..255

  const float p = probs[b * LF + t];
  float a;
  if (s < SS - 1) {
    const unsigned key = (unsigned)((s * BB + b) * LF + t) * 2654435761u + 0x9e3779b9u;
    a = (hash_uniform(key) < p) ? 1.0f : 0.0f;
  } else {
    a = (p >= 0.5f) ? 1.0f : 0.0f;
  }
  actions[(s * BB + b) * LF + t] = a;

  // Bernoulli log-prob term: a*log(p) + (1-a)*log1p(-p), via hw log2.
  const float lterm = 0.69314718056f *
      ((a > 0.5f) ? __log2f(p) : __log2f(1.0f - p));
  float lp = lterm;

#pragma unroll
  for (int off = 16; off > 0; off >>= 1) lp += __shfl_down(lp, off, 32);
  __shared__ float wsum[8];
  if ((t & 31) == 0) wsum[t >> 5] = lp;
  __syncthreads();
  if (t < 8) {
    float v = wsum[t];
#pragma unroll
    for (int off = 4; off > 0; off >>= 1) v += __shfl_down(v, off, 8);
    if (t == 0) logprobs[s * BB + b] = v;
  }
}

// ---------------------------------------------------------------------------
// Kernel 3 (bandwidth-dominant): one pass over f_logits computing all 4
// sample maxima per (b,v) column. TDM path: wave 0 issues double-buffered
// 16x256 f32 tiles (16 KB) global->LDS; all 8 waves consume from LDS.
// Per-element cost ~4 VALU + 1 TRANS + 8 gate ops -> below the HBM floor.
// ---------------------------------------------------------------------------
__global__ __launch_bounds__(256) void values_kernel(
    const float* __restrict__ q_logits,   // [B,LQ,V]
    const float* __restrict__ f_logits,   // [B,LF,V]
    const float* __restrict__ q_mask,     // [B,LQ]
    const float* __restrict__ amask,      // [B,LF]
    const float* __restrict__ actions,    // [S,B,LF]
    float* __restrict__ values)           // [S,B,V]
{
  const int b = blockIdx.y;
  const int t = threadIdx.x;
  const int v = blockIdx.x * 256 + t;     // 125*256 == 32000 exactly

  __shared__ float gate[SS][LF];
#pragma unroll
  for (int s = 0; s < SS; ++s)
    gate[s][t] = actions[(s * BB + b) * LF + t] * amask[b * LF + t];
  __syncthreads();

  // q_max over LQ rows (q_mask is a per-row scalar broadcast).
  float qm = -__builtin_inff();
#pragma unroll 4
  for (int lq = 0; lq < LQ; ++lq) {
    const float x = q_logits[(b * LQ + lq) * VV + v];
    qm = fmaxf(qm, log1p_relu_fast(x) * q_mask[b * LQ + lq]);
  }

  float m0 = qm, m1 = qm, m2 = qm, m3 = qm;

#if defined(HAVE_TDM)
  __shared__ float fbuf[2][16][256];                 // 32 KB double buffer
  const bool issuer = (t < 32);                      // wave 0 only (TDM ignores EXEC)
  const float* gtile = f_logits + (size_t)b * LF * VV + (size_t)blockIdx.x * 256;

  auto consume = [&](const float (*rows)[256], int lbase) {
#pragma unroll
    for (int r = 0; r < 16; ++r) {
      const int l = lbase + r;
      const float y = log1p_relu_fast(rows[r][t]);
      m0 = fmaxf(m0, y * gate[0][l]);
      m1 = fmaxf(m1, y * gate[1][l]);
      m2 = fmaxf(m2, y * gate[2][l]);
      m3 = fmaxf(m3, y * gate[3][l]);
    }
  };

  if (issuer) tdm_load_tile(gtile, lds_off(&fbuf[0][0][0]), 16);
#pragma unroll 1
  for (int tt = 0; tt < 15; ++tt) {
    if (issuer) {
      tdm_load_tile(gtile + (size_t)(tt + 1) * 16 * VV,
                    lds_off(&fbuf[(tt + 1) & 1][0][0]), 16);
      __builtin_amdgcn_s_wait_tensorcnt(1);          // tile tt landed (in-order)
    }
    __syncthreads();                                 // tile tt visible to all waves
    consume(fbuf[tt & 1], tt * 16);
    __syncthreads();                                 // done reading before reuse
  }
  if (issuer) __builtin_amdgcn_s_wait_tensorcnt(0);  // last tile landed
  __syncthreads();
  consume(fbuf[1], 240);
#else
  const float* fcol = f_logits + (size_t)b * LF * VV + v;
#pragma unroll 4
  for (int l = 0; l < LF; ++l) {
    __builtin_prefetch(fcol + (size_t)(l + 8) * VV, 0, 0);
    const float y = log1p_relu_fast(fcol[(size_t)l * VV]);
    m0 = fmaxf(m0, y * gate[0][l]);
    m1 = fmaxf(m1, y * gate[1][l]);
    m2 = fmaxf(m2, y * gate[2][l]);
    m3 = fmaxf(m3, y * gate[3][l]);
  }
#endif

  values[(0 * BB + b) * VV + v] = m0;
  values[(1 * BB + b) * VV + v] = m1;
  values[(2 * BB + b) * VV + v] = m2;
  values[(3 * BB + b) * VV + v] = m3;
}

// ---------------------------------------------------------------------------
// Launch
// ---------------------------------------------------------------------------
extern "C" void kernel_launch(void* const* d_in, const int* in_sizes, int n_in,
                              void* d_out, int out_size, void* d_ws, size_t ws_size,
                              hipStream_t stream) {
  const float* scores   = (const float*)d_in[0];  // [B,H,LQ,LF]
  const float* q_logits = (const float*)d_in[1];  // [B,LQ,V]
  const float* f_logits = (const float*)d_in[2];  // [B,LF,V]
  const float* q_mask   = (const float*)d_in[3];  // [B,LQ]
  const float* amask    = (const float*)d_in[4];  // [B,LF]
  (void)in_sizes; (void)n_in; (void)out_size; (void)ws_size;

  float* out      = (float*)d_out;
  float* values   = out;                                   // S*B*V = 1,024,000
  float* logprobs = out + (size_t)SS * BB * VV;            // 32
  float* actions  = logprobs + (size_t)SS * BB;            // S*B*LF = 8192
  float* probs    = (float*)d_ws;                          // [B,LF] scratch

  probs_wmma_kernel<<<dim3(BB * (LF / 16)), dim3(32), 0, stream>>>(scores, amask, probs);
  sample_kernel<<<dim3(SS, BB), dim3(256), 0, stream>>>(probs, actions, logprobs);
  values_kernel<<<dim3(VV / 256, BB), dim3(256), 0, stream>>>(
      q_logits, f_logits, q_mask, amask, actions, values);
}
